// TrajectoryDecoder_65764539236637
// MI455X (gfx1250) — matrix operand order repaired and checked by reading, not compiled, not run on hardware
//
#include <hip/hip_runtime.h>

// ---------------------------------------------------------------------------
// MI455X (gfx1250) fused 2-layer LSTM rollout, bf16 WMMA, persistent tiles.
// Batch=32768, HIDDEN=512, OUT=2, STEPS=60.
// One workgroup (256 thr = 8 wave32) owns 32 batch rows for all 60 steps
// (two 16-row WMMA M-tiles -> each streamed weight block feeds 2 WMMAs,
//  halving L2 weight traffic vs an M=16 tile).
// State lives in LDS (~258 KB; CDNA5 WGP has 320 KB).
// ---------------------------------------------------------------------------

typedef __attribute__((ext_vector_type(16))) __bf16 v16bf;
typedef __attribute__((ext_vector_type(8)))  float  v8f;

#define HID 512
#define STEPS 60
#define KC_CNT 16            // 512 / 32 K-chunks
#define MB 32                // batch rows per workgroup (2 WMMA M-tiles)
// packed weight offsets in ushort units inside d_ws
#define OFF_WIH   0u         // w_init_h packed: 512x512
#define OFF_WIC   262144u    // w_init_c packed: 512x512
#define OFF_WHH0  524288u    // w_hh0  packed: 2048x512 (gate-interleaved)
#define OFF_WIH1  1572864u   // w_ih1  packed
#define OFF_WHH1  2621440u   // w_hh1  packed
#define WS_NEEDED (7340032u) // bytes

__device__ __forceinline__ unsigned short f2bf(float f) {
  unsigned int u = __float_as_uint(f);
  u += 0x7FFFu + ((u >> 16) & 1u);          // round-to-nearest-even
  return (unsigned short)(u >> 16);
}
__device__ __forceinline__ float bf2f(unsigned short s) {
  return __uint_as_float(((unsigned int)s) << 16);
}

union V16U { v16bf v; uint4 q[2]; };

__device__ __forceinline__ v16bf ldv(const unsigned short* p) {
  V16U u;
  u.q[0] = *(const uint4*)(p);
  u.q[1] = *(const uint4*)(p + 8);
  return u.v;
}

__device__ __forceinline__ float sigf(float x) {
  return __builtin_amdgcn_rcpf(1.0f + __expf(-x));
}
__device__ __forceinline__ float tanh_f(float x) {
  float e = __expf(-2.0f * x);
  return (1.0f - e) * __builtin_amdgcn_rcpf(1.0f + e);
}

// Element (m,k) of a 32x512 bf16 matrix kept as two stacked 16-row WMMA
// A-operand tiles (8192 ushorts each). Within a tile: chunk kc holds
// K=kc*32..+31; lane l<16: M=l, halves 0..7 -> K 0..7, 8..15 -> K 16..23;
// lane l>=16: M=l-16, halves 0..7 -> K 8..15, 8..15 -> K 24..31.
__device__ __forceinline__ int hA_idx(int m, int k) {
  int kc = k >> 5, ko = k & 31, grp = ko >> 3;
  int laneA = (m & 15) + ((grp & 1) << 4);
  int half  = (ko & 7) + ((grp >> 1) << 3);
  return (m >> 4) * 8192 + kc * 512 + laneA * 16 + half;
}
__device__ __forceinline__ void store_hA(unsigned short* buf, int m, int k, float v) {
  buf[hA_idx(m, k)] = f2bf(v);
}
__device__ __forceinline__ float load_hA(const unsigned short* buf, int m, int k) {
  return bf2f(buf[hA_idx(m, k)]);
}

// ---------------------------------------------------------------------------
// Pack fp32 weight [N x 512] into bf16 WMMA B-operand blocks.
// Block (nt,kc) = 32x16 tile: B[k,n] = W[n_orig, k].
// In-block: lane l (0..31), 16 halves; h=l/16: hf<8 -> K=kc*32+8h+hf,
// hf>=8 -> K=kc*32+16+8h+(hf-8); N = nt*16 + l%16.
// interleave=1: packed row n' = 4*hidden + gate  (n_orig = (n'%4)*512 + n'/4)
// ---------------------------------------------------------------------------
__global__ void pack_kernel(const float* __restrict__ src,
                            unsigned short* __restrict__ dst,
                            int NT, int interleave) {
  int tid = blockIdx.x * blockDim.x + threadIdx.x;
  int total = NT * KC_CNT * 512;
  if (tid >= total) return;
  int blk = tid >> 9;           // (nt*16 + kc)
  int within = tid & 511;
  int lane = within >> 4;
  int hf = within & 15;
  int nt = blk >> 4;
  int kc = blk & 15;
  int npk = nt * 16 + (lane & 15);
  int k = kc * 32 + ((hf >> 3) << 4) + ((lane >> 4) << 3) + (hf & 7);
  int n = interleave ? ((npk & 3) * HID + (npk >> 2)) : npk;
  dst[tid] = f2bf(src[(size_t)n * HID + k]);
}

// LSTM cell epilogue for one 16-wide gate-interleaved N-tile, both M-tiles.
// Quad ds_swizzle broadcast fuses i,f,g,o per hidden unit in registers.
__device__ __forceinline__ void cell_post2(
    const v8f& acc0, const v8f& acc1, int nt, int lane,
    const float* __restrict__ bA, const float* __restrict__ bB,
    const float* __restrict__ wx,       // 2048x2 input weights, or nullptr
    const float* xrow,                  // LDS [MB][2] inputs, or nullptr
    float* cstate, unsigned short* hdst) {
  int npk = nt * 16 + (lane & 15);
  int q = npk >> 2;                     // hidden unit
  int g = lane & 3;                     // gate id: 0=i 1=f 2=g 3=o
  int grow = g * HID + q;
  float bias = bA[grow] + bB[grow];
  float wx0 = 0.f, wx1 = 0.f;
  if (wx) { wx0 = wx[grow * 2]; wx1 = wx[grow * 2 + 1]; }
#pragma unroll
  for (int mt = 0; mt < 2; ++mt) {
#pragma unroll
    for (int r = 0; r < 8; ++r) {
      int m = mt * 16 + r + ((lane >> 4) << 3);
      float pre = (mt ? acc1[r] : acc0[r]) + bias;
      if (wx) pre += xrow[m * 2] * wx0 + xrow[m * 2 + 1] * wx1;
      int pi = __float_as_int(pre);
      // group-of-4 ds_swizzle broadcasts: gather all four gates per lane
      float iv = __int_as_float(__builtin_amdgcn_ds_swizzle(pi, 0x8000));
      float fv = __int_as_float(__builtin_amdgcn_ds_swizzle(pi, 0x8055));
      float gv = __int_as_float(__builtin_amdgcn_ds_swizzle(pi, 0x80AA));
      float ov = __int_as_float(__builtin_amdgcn_ds_swizzle(pi, 0x80FF));
      float co = cstate[m * HID + q];
      float cn = sigf(fv) * co + sigf(iv) * tanh_f(gv);
      float hn = sigf(ov) * tanh_f(cn);
      if (g == 0) {
        cstate[m * HID + q] = cn;
        store_hA(hdst, m, q, hn);
      }
    }
  }
}

__global__ void __launch_bounds__(256)
traj_kernel(const float* __restrict__ h_in,
            const float* __restrict__ b_init_h, const float* __restrict__ b_init_c,
            const float* __restrict__ w_ih0,
            const float* __restrict__ b_ih0, const float* __restrict__ b_hh0,
            const float* __restrict__ b_ih1, const float* __restrict__ b_hh1,
            const float* __restrict__ w_out, const float* __restrict__ b_out,
            const unsigned short* __restrict__ wp,
            float* __restrict__ out) {
  __shared__ __align__(16) unsigned short h1buf[2][16384]; // 32x512 bf16, A-layout
  __shared__ __align__(16) unsigned short h2buf[2][16384];
  __shared__ float c1s[16384];                             // 32x512 fp32
  __shared__ float c2s[16384];
  __shared__ float xbuf[64];                               // [32][2] fed-back pred
  __shared__ float pp[512];                                // pred partials [32][8][2]

  const int tid = threadIdx.x;
  const int lane = tid & 31;
  const int w = tid >> 5;                 // wave id 0..7
  const int b0 = blockIdx.x * MB;

  const unsigned short* Wih  = wp + OFF_WIH;
  const unsigned short* Wic  = wp + OFF_WIC;
  const unsigned short* Whh0 = wp + OFF_WHH0;
  const unsigned short* Wih1 = wp + OFF_WIH1;
  const unsigned short* Whh1 = wp + OFF_WHH1;

  const v8f vzero = {0.f, 0.f, 0.f, 0.f, 0.f, 0.f, 0.f, 0.f};

  // ---- stage input rows into A-layout bf16 (use h1buf[1] as scratch) ----
  for (int i = 0; i < 64; ++i) {
    int e = i * 256 + tid;                // 32*512 = 16384 elements
    int m = e >> 9, k = e & 511;
    store_hA(h1buf[1], m, k, h_in[(size_t)(b0 + m) * HID + k]);
  }
  __syncthreads();

  // ---- init GEMMs: h0 = h @ Wih^T + b, c0 = h @ Wic^T + b (N=512) -------
#pragma unroll 1
  for (int j = 0; j < 4; ++j) {
    int nt = w * 4 + j;
    v8f aH0 = vzero, aH1 = vzero, aC0 = vzero, aC1 = vzero;
    for (int kc = 0; kc < KC_CNT; ++kc) {
      v16bf A0 = ldv(&h1buf[1][kc * 512 + lane * 16]);
      v16bf A1 = ldv(&h1buf[1][8192 + kc * 512 + lane * 16]);
      v16bf Bh = ldv(Wih + (size_t)(nt * 16 + kc) * 512 + lane * 16);
      aH0 = __builtin_amdgcn_wmma_f32_16x16x32_bf16(false, A0, false, Bh, (short)0, aH0, false, false);
      aH1 = __builtin_amdgcn_wmma_f32_16x16x32_bf16(false, A1, false, Bh, (short)0, aH1, false, false);
      v16bf Bc = ldv(Wic + (size_t)(nt * 16 + kc) * 512 + lane * 16);
      aC0 = __builtin_amdgcn_wmma_f32_16x16x32_bf16(false, A0, false, Bc, (short)0, aC0, false, false);
      aC1 = __builtin_amdgcn_wmma_f32_16x16x32_bf16(false, A1, false, Bc, (short)0, aC1, false, false);
    }
    int n = nt * 16 + (lane & 15);
    float bh = b_init_h[n], bc = b_init_c[n];
#pragma unroll
    for (int mt = 0; mt < 2; ++mt) {
#pragma unroll
      for (int r = 0; r < 8; ++r) {
        int m = mt * 16 + r + ((lane >> 4) << 3);
        float hv = (mt ? aH1[r] : aH0[r]) + bh;
        float cv = (mt ? aC1[r] : aC0[r]) + bc;
        store_hA(h1buf[0], m, n, hv);
        store_hA(h2buf[0], m, n, hv);
        c1s[m * HID + n] = cv;
        c2s[m * HID + n] = cv;
      }
    }
  }
  if (tid < 64) xbuf[tid] = 0.0f;
  __syncthreads();

  // ------------------------------- rollout --------------------------------
  int pb = 0;
#pragma unroll 1
  for (int t = 0; t < STEPS; ++t) {
    // ---- layer 0: gates = h1 @ Whh0^T (+x rank-2 +bias in epilogue) ------
#pragma unroll 1
    for (int nb = 0; nb < 4; ++nb) {
      v8f acc0[4] = {vzero, vzero, vzero, vzero};
      v8f acc1[4] = {vzero, vzero, vzero, vzero};
      for (int kc = 0; kc < KC_CNT; ++kc) {
        v16bf A0 = ldv(&h1buf[pb][kc * 512 + lane * 16]);
        v16bf A1 = ldv(&h1buf[pb][8192 + kc * 512 + lane * 16]);
#pragma unroll
        for (int j = 0; j < 4; ++j) {
          int nt = w * 16 + nb * 4 + j;
          v16bf B = ldv(Whh0 + (size_t)(nt * 16 + kc) * 512 + lane * 16);
          acc0[j] = __builtin_amdgcn_wmma_f32_16x16x32_bf16(false, A0, false, B, (short)0, acc0[j], false, false);
          acc1[j] = __builtin_amdgcn_wmma_f32_16x16x32_bf16(false, A1, false, B, (short)0, acc1[j], false, false);
        }
      }
#pragma unroll
      for (int j = 0; j < 4; ++j)
        cell_post2(acc0[j], acc1[j], w * 16 + nb * 4 + j, lane,
                   b_ih0, b_hh0, w_ih0, xbuf, c1s, h1buf[pb ^ 1]);
    }
    __syncthreads();

    // ---- layer 1: gates = h1_new @ Wih1^T + h2 @ Whh1^T ------------------
#pragma unroll 1
    for (int nb = 0; nb < 4; ++nb) {
      v8f acc0[4] = {vzero, vzero, vzero, vzero};
      v8f acc1[4] = {vzero, vzero, vzero, vzero};
      for (int kc = 0; kc < KC_CNT; ++kc) {
        v16bf A10 = ldv(&h1buf[pb ^ 1][kc * 512 + lane * 16]);
        v16bf A11 = ldv(&h1buf[pb ^ 1][8192 + kc * 512 + lane * 16]);
        v16bf A20 = ldv(&h2buf[pb][kc * 512 + lane * 16]);
        v16bf A21 = ldv(&h2buf[pb][8192 + kc * 512 + lane * 16]);
#pragma unroll
        for (int j = 0; j < 4; ++j) {
          int nt = w * 16 + nb * 4 + j;
          v16bf B1 = ldv(Wih1 + (size_t)(nt * 16 + kc) * 512 + lane * 16);
          acc0[j] = __builtin_amdgcn_wmma_f32_16x16x32_bf16(false, A10, false, B1, (short)0, acc0[j], false, false);
          acc1[j] = __builtin_amdgcn_wmma_f32_16x16x32_bf16(false, A11, false, B1, (short)0, acc1[j], false, false);
          v16bf B2 = ldv(Whh1 + (size_t)(nt * 16 + kc) * 512 + lane * 16);
          acc0[j] = __builtin_amdgcn_wmma_f32_16x16x32_bf16(false, A20, false, B2, (short)0, acc0[j], false, false);
          acc1[j] = __builtin_amdgcn_wmma_f32_16x16x32_bf16(false, A21, false, B2, (short)0, acc1[j], false, false);
        }
      }
#pragma unroll
      for (int j = 0; j < 4; ++j)
        cell_post2(acc0[j], acc1[j], w * 16 + nb * 4 + j, lane,
                   b_ih1, b_hh1, nullptr, nullptr, c2s, h2buf[pb ^ 1]);
    }
    __syncthreads();

    // ---- pred = h2 @ w_out^T + b_out (rank-2), feed back as next x -------
    {
      int m = tid >> 3, p = tid & 7;
      float s0 = 0.f, s1 = 0.f;
      const unsigned short* hb = h2buf[pb ^ 1];
      for (int k = p * 64; k < p * 64 + 64; ++k) {
        float hv = load_hA(hb, m, k);
        s0 += hv * w_out[k];
        s1 += hv * w_out[HID + k];
      }
      pp[(m * 8 + p) * 2 + 0] = s0;
      pp[(m * 8 + p) * 2 + 1] = s1;
    }
    __syncthreads();
    if (tid < 64) {
      int m = tid >> 1, j = tid & 1;
      float s = b_out[j];
#pragma unroll
      for (int p = 0; p < 8; ++p) s += pp[(m * 8 + p) * 2 + j];
      out[((size_t)(b0 + m) * STEPS + t) * 2 + j] = s;
      xbuf[m * 2 + j] = s;
    }
    __syncthreads();
    pb ^= 1;
  }
}

extern "C" void kernel_launch(void* const* d_in, const int* in_sizes, int n_in,
                              void* d_out, int out_size, void* d_ws, size_t ws_size,
                              hipStream_t stream) {
  (void)in_sizes; (void)n_in; (void)out_size;
  if (ws_size < WS_NEEDED) return;

  const float* h        = (const float*)d_in[0];
  const float* w_init_h = (const float*)d_in[1];
  const float* b_init_h = (const float*)d_in[2];
  const float* w_init_c = (const float*)d_in[3];
  const float* b_init_c = (const float*)d_in[4];
  const float* w_ih0    = (const float*)d_in[5];
  const float* w_hh0    = (const float*)d_in[6];
  const float* b_ih0    = (const float*)d_in[7];
  const float* b_hh0    = (const float*)d_in[8];
  const float* w_ih1    = (const float*)d_in[9];
  const float* w_hh1    = (const float*)d_in[10];
  const float* b_ih1    = (const float*)d_in[11];
  const float* b_hh1    = (const float*)d_in[12];
  const float* w_out    = (const float*)d_in[13];
  const float* b_out    = (const float*)d_in[14];
  unsigned short* wp = (unsigned short*)d_ws;
  float* out = (float*)d_out;

  // pack weights to bf16 WMMA-B block layout (gate rows interleaved for LSTM)
  pack_kernel<<<(32 * KC_CNT * 512) / 256, 256, 0, stream>>>(w_init_h, wp + OFF_WIH, 32, 0);
  pack_kernel<<<(32 * KC_CNT * 512) / 256, 256, 0, stream>>>(w_init_c, wp + OFF_WIC, 32, 0);
  pack_kernel<<<(128 * KC_CNT * 512) / 256, 256, 0, stream>>>(w_hh0, wp + OFF_WHH0, 128, 1);
  pack_kernel<<<(128 * KC_CNT * 512) / 256, 256, 0, stream>>>(w_ih1, wp + OFF_WIH1, 128, 1);
  pack_kernel<<<(128 * KC_CNT * 512) / 256, 256, 0, stream>>>(w_hh1, wp + OFF_WHH1, 128, 1);

  // persistent fused rollout: 32768/32 = 1024 tiles
  traj_kernel<<<1024, 256, 0, stream>>>(h, b_init_h, b_init_c,
                                        w_ih0, b_ih0, b_hh0,
                                        b_ih1, b_hh1,
                                        w_out, b_out, wp, out);
}
